// RingDilatedAttentionProduction_23029614641315
// MI455X (gfx1250) — matrix operand order repaired and checked by reading, not compiled, not run on hardware
//
#include <hip/hip_runtime.h>
#include <stdint.h>

// ---------------------------------------------------------------------------
// Dilated flash attention for MI455X (gfx1250, wave32, v_wmma_f32_16x16x32_bf16).
// b=1, n=8192, h=8, d=64. Units = (group, head, segment):
//   group0: heads 0-1, s=2048, r=1, off=0  (8 units)
//   group1: heads 2-4, s=4096, r=2, off=1  (6 units)
//   group2: heads 5-7, s=8192, r=4, off=2  (3 units)
// Gather indices repeat with period m = s/r = 2048 and the output scatter keeps
// only the last repetition -> 2048 visible query rows per unit, attending over
// all s gathered (duplicated) key columns. Pre-pass packs Q/K (row-major) and
// V (transposed) as bf16 with the gather applied, so the hot loop does zero
// format conversion and loads all fragments as b128.
// Softmax runs in the exp2 domain (log2e folded into the score scale) and the
// flash rescale is skipped via a wave-uniform ballot when no row max changed.
// ---------------------------------------------------------------------------

typedef __attribute__((ext_vector_type(16))) __bf16       v16bf;
typedef __attribute__((ext_vector_type(8)))  float        v8f;
typedef __attribute__((ext_vector_type(4)))  unsigned int u32x4;
typedef __attribute__((ext_vector_type(4)))  float        f32x4;

#define HEADS  8
#define DIM    64
#define NUNITS 17
#define MROWS  2048   // gather period == visible rows per unit

struct Unit { int head, seg_start, rsh, offset, j0v; };

__device__ __forceinline__ Unit decode_unit(int u) {
  Unit t;
  if (u < 8)       { t.head = u >> 2;                      t.seg_start = (u & 3) * 2048; t.rsh = 0; t.offset = 0; t.j0v = 0;    }
  else if (u < 14) { int v2 = u - 8; t.head = 2 + (v2 >> 1); t.seg_start = (v2 & 1) * 4096; t.rsh = 1; t.offset = 1; t.j0v = 2048; }
  else             { t.head = 5 + (u - 14);                t.seg_start = 0;              t.rsh = 2; t.offset = 2; t.j0v = 6144; }
  return t;
}

// two f32 -> packed bf16 pair (truncation) with one v_perm_b32
__device__ __forceinline__ unsigned int pack_bf16(float lo, float hi) {
  return __builtin_amdgcn_perm(__float_as_uint(hi), __float_as_uint(lo), 0x07060302u);
}

__global__ void zero_out_kernel(float* __restrict__ out, int n) {
  int i = (blockIdx.x * blockDim.x + threadIdx.x) * 4;
  if (i + 3 < n) {
    f32x4 z = {0.f, 0.f, 0.f, 0.f};
    *(f32x4*)(out + i) = z;
  }
}

// One thread per (unit, kidx): convert+gather one Q row, one K row (row-major
// bf16) and one V row (scattered into the transposed [unit][d][kidx] buffer).
__global__ __launch_bounds__(256) void prepack_kernel(
    const float* __restrict__ Q, const float* __restrict__ K,
    const float* __restrict__ V,
    __bf16* __restrict__ Qg, __bf16* __restrict__ Kg, __bf16* __restrict__ Vt) {
  const int idx = blockIdx.x * blockDim.x + threadIdx.x;
  const int u  = idx >> 11;
  const int ki = idx & (MROWS - 1);
  if (u >= NUNITS) return;
  Unit t = decode_unit(u);
  const int pos = t.seg_start + t.offset + (ki << t.rsh);
  const size_t src = ((size_t)pos * HEADS + t.head) * DIM;
  const size_t dstrow = ((size_t)u * MROWS + ki) * DIM;
  const f32x4* qr = (const f32x4*)(Q + src);
  const f32x4* kr = (const f32x4*)(K + src);
  const f32x4* vr = (const f32x4*)(V + src);
  u32x4* qo = (u32x4*)(Qg + dstrow);
  u32x4* ko = (u32x4*)(Kg + dstrow);
  unsigned short* vo = (unsigned short*)Vt + (size_t)u * (DIM * MROWS) + ki;
#pragma unroll
  for (int c = 0; c < 8; ++c) {
    f32x4 a = qr[2 * c], b = qr[2 * c + 1];
    u32x4 o = { pack_bf16(a.x, a.y), pack_bf16(a.z, a.w),
                pack_bf16(b.x, b.y), pack_bf16(b.z, b.w) };
    qo[c] = o;
    f32x4 e = kr[2 * c], f = kr[2 * c + 1];
    u32x4 o2 = { pack_bf16(e.x, e.y), pack_bf16(e.z, e.w),
                 pack_bf16(f.x, f.y), pack_bf16(f.z, f.w) };
    ko[c] = o2;
  }
#pragma unroll
  for (int g = 0; g < 16; ++g) {
    f32x4 a = vr[g];
    vo[(size_t)(4 * g + 0) * MROWS] = (unsigned short)(__float_as_uint(a.x) >> 16);
    vo[(size_t)(4 * g + 1) * MROWS] = (unsigned short)(__float_as_uint(a.y) >> 16);
    vo[(size_t)(4 * g + 2) * MROWS] = (unsigned short)(__float_as_uint(a.z) >> 16);
    vo[(size_t)(4 * g + 3) * MROWS] = (unsigned short)(__float_as_uint(a.w) >> 16);
  }
}

__global__ __launch_bounds__(256) void dilated_attn_kernel(
    const __bf16* __restrict__ Qg, const __bf16* __restrict__ Kg,
    const __bf16* __restrict__ Vt, const int* __restrict__ causal_flag,
    float* __restrict__ O) {
  const int lane = threadIdx.x & 31;
  const int w    = threadIdx.x >> 5;   // wave in block (0..7)
  const int lo   = lane & 15;
  const int hi   = lane >> 4;

  const int u  = blockIdx.x >> 4;
  const int tb = blockIdx.x & 15;
  Unit t = decode_unit(u);
  const int causal = __builtin_amdgcn_readfirstlane(*causal_flag);

  const int q0  = t.j0v + tb * 128 + w * 16;  // first gathered q row of wave
  const int vr0 = q0 - t.j0v;                 // visible-row base within unit

  __shared__ __align__(16) float spad[8][16 * 32];
  __shared__ __align__(16) float apad[8][16];
  float* sp = spad[w];
  float* ap = apad[w];

  union V16 { u32x4 q[2]; v16bf v; };

  // ---- Q fragments (A layout), lane row = lo, bf16 direct loads ----
  const __bf16* qg = Qg + ((size_t)u * MROWS + (vr0 + lo)) * DIM;
  V16 qa[2];
#pragma unroll
  for (int c = 0; c < 2; ++c) {
    qa[c].q[0] = *(const u32x4*)(qg + c * 32 + 8 * hi);        // K = 8*hi+e
    qa[c].q[1] = *(const u32x4*)(qg + c * 32 + 16 + 8 * hi);   // K = 16+8*hi+e
  }

  const __bf16* kgu = Kg + (size_t)u * ((size_t)MROWS * DIM);
  const __bf16* vtu = Vt + (size_t)u * ((size_t)DIM * MROWS);
  const __bf16* vrow[4];
#pragma unroll
  for (int d = 0; d < 4; ++d)
    vrow[d] = vtu + (size_t)(d * 16 + lo) * MROWS + 8 * hi;    // B lane = d col

  v8f acc[4];
#pragma unroll
  for (int d = 0; d < 4; ++d) acc[d] = (v8f){};
  float m_prev = -1e30f, lsum = 0.f;
  // 1/sqrt(64) * log2(e): softmax runs in the exp2 domain
  const float scale = 0.125f * 1.44269504088896f;

  const int kv_end = q0 + 16;
  for (int kv0 = 0; kv0 < kv_end; kv0 += 32) {
    // ---------------- S = Q * K^T (two 16x16 tiles) ----------------
    v8f sc[2];
#pragma unroll
    for (int n = 0; n < 2; ++n) {
      const int kidx = (kv0 + 16 * n + lo) & (MROWS - 1);      // B lane = key
      const __bf16* kr = kgu + (size_t)kidx * DIM + 8 * hi;    // B: K = 8*hi+e, 16+8*hi+e
      v8f ctile = (v8f){};
#pragma unroll
      for (int c = 0; c < 2; ++c) {
        V16 kb;
        kb.q[0] = *(const u32x4*)(kr + c * 32);
        kb.q[1] = *(const u32x4*)(kr + c * 32 + 16);
        ctile = __builtin_amdgcn_wmma_f32_16x16x32_bf16(
            false, qa[c].v, false, kb.v, (short)0, ctile, false, false);
      }
      sc[n] = ctile;
    }

    // scale, mask only on the single partially-causal step (uniform guard)
    if (causal && (kv0 + 31 > q0)) {
#pragma unroll
      for (int r = 0; r < 8; ++r) {
        const int row = q0 + r + 8 * hi;
        float s0 = sc[0][r] * scale;
        float s1 = sc[1][r] * scale;
        if (kv0 + lo > row)      s0 = -1e30f;
        if (kv0 + 16 + lo > row) s1 = -1e30f;
        sc[0][r] = s0; sc[1][r] = s1;
      }
    } else {
#pragma unroll
      for (int r = 0; r < 8; ++r) { sc[0][r] *= scale; sc[1][r] *= scale; }
    }

    // ------- C layout -> row-per-lane (A-element order) via LDS -------
#pragma unroll
    for (int r = 0; r < 8; ++r) {
      sp[(r + 8 * hi) * 32 + lo]      = sc[0][r];
      sp[(r + 8 * hi) * 32 + 16 + lo] = sc[1][r];
    }
    __builtin_amdgcn_wave_barrier();
    union F16x { f32x4 f4[4]; float f[16]; } sv;
    const float* srow = sp + lo * 32;                // row = q0 + lo
    sv.f4[0] = *(const f32x4*)(srow + 8 * hi);       // cols 8*hi .. +7
    sv.f4[1] = *(const f32x4*)(srow + 8 * hi + 4);
    sv.f4[2] = *(const f32x4*)(srow + 16 + 8 * hi);  // cols 16+8*hi .. +7
    sv.f4[3] = *(const f32x4*)(srow + 16 + 8 * hi + 4);
    __builtin_amdgcn_wave_barrier();

    // ---------------- online softmax (exp2 domain), in-lane reduction ----
    float tmax = sv.f[0];
#pragma unroll
    for (int e = 1; e < 16; ++e) tmax = fmaxf(tmax, sv.f[e]);
    tmax = fmaxf(tmax, __shfl_xor(tmax, 16, 32));    // combine hi halves
    const float mnew = fmaxf(m_prev, tmax);

    // rescale only when some row's max actually changed (wave-uniform branch)
    if (__builtin_amdgcn_ballot_w32(mnew != m_prev) != 0u) {
      const float alpha = __builtin_amdgcn_exp2f(m_prev - mnew);
      lsum *= alpha;
      ap[lo] = alpha;
      __builtin_amdgcn_wave_barrier();
      f32x4 al0 = *(const f32x4*)(ap + 8 * hi);
      f32x4 al1 = *(const f32x4*)(ap + 8 * hi + 4);
      __builtin_amdgcn_wave_barrier();
      const float alc[8] = { al0.x, al0.y, al0.z, al0.w, al1.x, al1.y, al1.z, al1.w };
#pragma unroll
      for (int d = 0; d < 4; ++d)
#pragma unroll
        for (int r = 0; r < 8; ++r) acc[d][r] *= alc[r];
    }
    m_prev = mnew;

    float p[16];
    float rs = 0.f;
#pragma unroll
    for (int e = 0; e < 16; ++e) {
      p[e] = __builtin_amdgcn_exp2f(sv.f[e] - mnew);
      rs += p[e];
    }
    rs += __shfl_xor(rs, 16, 32);
    lsum += rs;

    // ---- P is already in A-element order: pack f32 -> bf16 (8 v_perm) ----
    V16 pa;
#pragma unroll
    for (int i = 0; i < 8; ++i)
      pa.q[i >> 2][i & 3] = pack_bf16(p[2 * i], p[2 * i + 1]);

    // ---------------- O += P * V (four 16-wide d tiles) ----------------
    const int kb0 = kv0 & (MROWS - 1);
#pragma unroll
    for (int d = 0; d < 4; ++d) {
      V16 vb;
      vb.q[0] = *(const u32x4*)(vrow[d] + kb0);      // keys kv0+8*hi+e, kv0+16+8*hi+e
      vb.q[1] = *(const u32x4*)(vrow[d] + kb0 + 16);
      acc[d] = __builtin_amdgcn_wmma_f32_16x16x32_bf16(
          false, pa.v, false, vb.v, (short)0, acc[d], false, false);
    }
  }

  // ---------------- finalize: broadcast l, normalize, scatter ----------------
  ap[lo] = lsum;
  __builtin_amdgcn_wave_barrier();
  f32x4 l0 = *(const f32x4*)(ap + 8 * hi);
  f32x4 l1 = *(const f32x4*)(ap + 8 * hi + 4);
  const float lc[8] = { l0.x, l0.y, l0.z, l0.w, l1.x, l1.y, l1.z, l1.w };
#pragma unroll
  for (int r = 0; r < 8; ++r) {
    const float inv = 1.0f / lc[r];
    const int j    = q0 + r + 8 * hi;
    const int opos = t.seg_start + t.offset + ((j - t.j0v) << t.rsh);
    float* orow = O + ((size_t)opos * HEADS + t.head) * DIM;
#pragma unroll
    for (int d = 0; d < 4; ++d) orow[d * 16 + lo] = acc[d][r] * inv;
  }
}

extern "C" void kernel_launch(void* const* d_in, const int* in_sizes, int n_in,
                              void* d_out, int out_size, void* d_ws, size_t ws_size,
                              hipStream_t stream) {
  const float* Q = (const float*)d_in[0];
  const float* K = (const float*)d_in[1];
  const float* V = (const float*)d_in[2];
  const int* causal = (const int*)d_in[3];
  float* O = (float*)d_out;

  // Scratch layout: Qg | Kg | Vt, each NUNITS*2048*64 bf16 (~4.25 MB each).
  const size_t per_bytes = (size_t)NUNITS * MROWS * DIM * 2;
  __bf16* Qg = (__bf16*)d_ws;
  __bf16* Kg = (__bf16*)((char*)d_ws + per_bytes);
  __bf16* Vt = (__bf16*)((char*)d_ws + 2 * per_bytes);

  // Zero output: dilated groups only scatter to gathered positions.
  const int zthreads = (out_size + 3) / 4;
  zero_out_kernel<<<(zthreads + 255) / 256, 256, 0, stream>>>(O, out_size);

  // Convert + gather + pack Q/K/V to bf16 (V transposed per unit).
  prepack_kernel<<<(NUNITS * MROWS) / 256, 256, 0, stream>>>(Q, K, V, Qg, Kg, Vt);

  // 17 units * 16 tiles of 128 visible query rows, 8 waves/block.
  dilated_attn_kernel<<<NUNITS * 16, 256, 0, stream>>>(Qg, Kg, Vt, causal, O);
}
